// NodeModel_ONE_20839181320242
// MI455X (gfx1250) — matrix operand order repaired and checked by reading, not compiled, not run on hardware
//
#include <hip/hip_runtime.h>
#include <hip/hip_bf16.h>

// MI455X (gfx1250) implementation.
// Roofline: edge GEMMs ~105 GFLOP dominate; HBM floor ~0.5 GB (edge_attr).
// -> compute-bound in fp32; use bf16 WMMA (v_wmma_f32_16x16x32_bf16, f32 acc).
// BN folded into second GEMM of each MLP after a stats pass (sum/sumsq atomics).
// bf16 tile staging (edge_mlp2 / node_mlp2) uses GLOBAL_LOAD_ASYNC_TO_LDS_B128
// via inline asm (ROCm clang-22 declares the builtin with LangAS::cuda_device
// params that are unspellable from HIP source), tracked with ASYNCcnt.

#define NNODES 100000
#define NEDGES 1600000
#define EPSV   1e-5f
#define SLOPE  0.01f

typedef __attribute__((ext_vector_type(16))) __bf16   v16bf;
typedef __attribute__((ext_vector_type(8)))  float    v8f;
typedef __attribute__((ext_vector_type(8)))  unsigned v8u;

// ---- async global -> LDS (CDNA5 ASYNCcnt path, inline asm) -----------------
// ISA: each lane copies 16 B from global VADDR (64b) to LDS byte address VDST; ASYNCcnt.
// Low 32 bits of a generic LDS address == LDS byte offset (aperture truncation).
static __device__ inline void async_copy_b128(const void* g, void* l) {
  unsigned lds = (unsigned)(unsigned long long)l;
  asm volatile("global_load_async_to_lds_b128 %0, %1, off"
               :: "v"(lds), "v"(g) : "memory");
}
static __device__ inline void wait_async0() {
#if __has_builtin(__builtin_amdgcn_s_wait_asynccnt)
  __builtin_amdgcn_s_wait_asynccnt(0);
#else
  asm volatile("s_wait_asynccnt 0x0" ::: "memory");
#endif
}
// ---------------------------------------------------------------------------

static __device__ inline unsigned short f2bf(float f) {
  unsigned u = __float_as_uint(f);
  u += ((u >> 16) & 1u) + 0x7fffu;       // round-to-nearest-even
  return (unsigned short)(u >> 16);
}
static __device__ inline unsigned pack2(float a, float b) {
  return (unsigned)f2bf(a) | ((unsigned)f2bf(b) << 16);
}
static __device__ inline v16bf mkfrag(uint4 lo, uint4 hi) {
  v8u t = (v8u){lo.x, lo.y, lo.z, lo.w, hi.x, hi.y, hi.z, hi.w};
  union { v8u u; v16bf b; } c; c.u = t; return c.b;
}
// A-matrix 16x32 bf16 fragment from LDS tile (row-major, strideU dwords/row).
// ISA layout: lane m=L&15, half=L>>4; VGPR j<4 -> K=2j,2j+1 (+8*half),
// VGPR j>=4 -> K=2j+8,2j+9 (+8*half). => per lane: dwords [4h..4h+3] and [8+4h..8+4h+3].
static __device__ inline v16bf load_a_frag(const unsigned* ldsA, int rowBase, int strideU, int kt) {
  int lane = threadIdx.x & 31;
  int m = lane & 15, half = lane >> 4;
  const unsigned* p = ldsA + (rowBase + m) * strideU + kt * 16 + half * 4;
  uint4 lo = *(const uint4*)p;
  uint4 hi = *(const uint4*)(p + 8);
  return mkfrag(lo, hi);
}
// B fragments are pre-swizzled per-lane-contiguous (32B/lane) in global ws.
static __device__ inline v16bf load_b_frag(const unsigned short* g) {
  int lane = threadIdx.x & 31;
  const uint4* p = (const uint4*)(g + lane * 16);
  return mkfrag(p[0], p[1]);
}
static __device__ inline v8f wmma_bf16(v16bf a, v16bf b, v8f c) {
  return __builtin_amdgcn_wmma_f32_16x16x32_bf16(false, a, false, b, (short)0, c, false, false);
}
// B-layout: lane L holds column n=L&15, elem i -> K = 32*kt + (i&7) + 8*(L>>4) + 16*(i>>3).
static __device__ inline size_t bfrag_off(int k, int j, int NT) {
  int kt = k >> 5, kl = k & 31;
  int i = (kl & 7) | ((kl >> 4) << 3), hi = (kl >> 3) & 1;
  int nt = j >> 4, n = j & 15;
  int lane = hi * 16 + n;
  return ((size_t)(kt * NT + nt) * 32 + lane) * 16 + i;
}

__global__ void zero_f32(float* p, long n) {
  long i = (long)blockIdx.x * blockDim.x + threadIdx.x;
  if (i < n) p[i] = 0.f;
}

// Convert W1 (128x128) and W3 (192x128) f32 -> bf16 fragment layout.
__global__ void prep_weights(const float* __restrict__ W1, const float* __restrict__ W3,
                             unsigned short* __restrict__ w1f, unsigned short* __restrict__ w3f) {
  int idx = blockIdx.x * 256 + threadIdx.x;
  if (idx < 128 * 128) {
    int k = idx >> 7, j = idx & 127;
    w1f[bfrag_off(k, j, 8)] = f2bf(W1[idx]);
  } else if (idx < 128 * 128 + 192 * 128) {
    int t = idx - 128 * 128;
    int k = t >> 7, j = t & 127;
    w3f[bfrag_off(k, j, 8)] = f2bf(W3[t]);
  }
}

// Fold BN (stats over `count` rows) into next weight matrix W (128 x Jdim):
// W'[k][j] = s_k * W[k][j], bias'[j] = b[j] + sum_k (beta_k - mean_k*s_k) * W[k][j]
__global__ void fold_bn(const float* __restrict__ stats, float count,
                        const float* __restrict__ gamma, const float* __restrict__ beta,
                        const float* __restrict__ W, const float* __restrict__ b,
                        int Jdim, int NT,
                        unsigned short* __restrict__ wf, float* __restrict__ biasp) {
  __shared__ float s_s[128], s_t[128];
  int j = threadIdx.x;
  {
    float mean = stats[j] / count;
    float var  = stats[128 + j] / count - mean * mean;
    float sc = gamma[j] * rsqrtf(var + EPSV);
    s_s[j] = sc;
    s_t[j] = beta[j] - mean * sc;
  }
  __syncthreads();
  if (j < Jdim) {
    float acc = b[j];
    for (int k = 0; k < 128; ++k) {
      float wv = W[k * Jdim + j];
      acc += s_t[k] * wv;
      wf[bfrag_off(k, j, NT)] = f2bf(s_s[k] * wv);
    }
    biasp[j] = acc;
  }
}

// Edge pass 1: h1 = leakyrelu([x[row] | edge_attr] @ W1 + b1); store bf16 + stats.
__global__ __launch_bounds__(256) void edge_mlp1(
    const float* __restrict__ x, const int* __restrict__ ei, const float* __restrict__ ea,
    const float* __restrict__ b1, const unsigned short* __restrict__ w1f,
    unsigned short* __restrict__ h1, float* __restrict__ stats) {
  __shared__ __align__(16) unsigned ldsA[128 * 68];
  int tid = threadIdx.x;
  long e0 = (long)blockIdx.x * 128;
  {
    int er = tid >> 1, hf = tid & 1;   // 2 threads/edge: hf0 = x gather, hf1 = edge_attr
    long e = e0 + er;
    const float* src = hf ? (ea + e * 64) : (x + (long)ei[e] * 64);
    unsigned* dst = ldsA + er * 68 + hf * 32;
#pragma unroll
    for (int i = 0; i < 16; ++i) {
      float4 v = ((const float4*)src)[i];
      dst[2 * i]     = pack2(v.x, v.y);
      dst[2 * i + 1] = pack2(v.z, v.w);
    }
    if (e + 128 < NEDGES) __builtin_prefetch(ea + (e + 128) * 64, 0, 0);
  }
  __syncthreads();
  int lane = tid & 31, nt = tid >> 5;
  int n = lane & 15, half = lane >> 4;
  int ch = nt * 16 + n;
  v16bf B0 = load_b_frag(w1f + (size_t)((0 * 8 + nt) * 32) * 16);
  v16bf B1 = load_b_frag(w1f + (size_t)((1 * 8 + nt) * 32) * 16);
  v16bf B2 = load_b_frag(w1f + (size_t)((2 * 8 + nt) * 32) * 16);
  v16bf B3 = load_b_frag(w1f + (size_t)((3 * 8 + nt) * 32) * 16);
  float bias = b1[ch];
  float ssum = 0.f, ssq = 0.f;
#pragma unroll
  for (int mt = 0; mt < 8; ++mt) {
    v8f c = {};
    c = wmma_bf16(load_a_frag(ldsA, mt * 16, 68, 0), B0, c);
    c = wmma_bf16(load_a_frag(ldsA, mt * 16, 68, 1), B1, c);
    c = wmma_bf16(load_a_frag(ldsA, mt * 16, 68, 2), B2, c);
    c = wmma_bf16(load_a_frag(ldsA, mt * 16, 68, 3), B3, c);
#pragma unroll
    for (int r = 0; r < 8; ++r) {
      float v = c[r] + bias;
      v = v > 0.f ? v : SLOPE * v;
      ssum += v; ssq += v * v;
      long row = e0 + mt * 16 + r + 8 * half;   // C layout: VGPR r -> M=r (+8 hi half)
      h1[row * 128 + ch] = f2bf(v);
    }
  }
  atomicAdd(&stats[ch], ssum);
  atomicAdd(&stats[128 + ch], ssq);
}

// Edge pass 2: h2 = h1 @ W2' + b2'; scatter-add into agg[col].
__global__ __launch_bounds__(256) void edge_mlp2(
    const unsigned short* __restrict__ h1, const int* __restrict__ ei,
    const unsigned short* __restrict__ w2f, const float* __restrict__ biasp,
    float* __restrict__ agg) {
  __shared__ __align__(16) unsigned ldsA[128 * 68];
  __shared__ int ldsCol[128];
  int tid = threadIdx.x;
  long e0 = (long)blockIdx.x * 128;
  if (tid < 128) ldsCol[tid] = ei[NEDGES + e0 + tid];
  {
    int er = tid >> 1, hf = tid & 1;
    const unsigned short* gsrc = h1 + (e0 + er) * 128 + hf * 64;
    unsigned* dst = ldsA + er * 68 + hf * 32;
#pragma unroll
    for (int i = 0; i < 8; ++i) async_copy_b128(gsrc + i * 8, dst + i * 4);
  }
  wait_async0();
  __syncthreads();
  int lane = tid & 31, nt = tid >> 5;
  int n = lane & 15, half = lane >> 4;
  int ch = nt * 16 + n;
  v16bf B0 = load_b_frag(w2f + (size_t)((0 * 8 + nt) * 32) * 16);
  v16bf B1 = load_b_frag(w2f + (size_t)((1 * 8 + nt) * 32) * 16);
  v16bf B2 = load_b_frag(w2f + (size_t)((2 * 8 + nt) * 32) * 16);
  v16bf B3 = load_b_frag(w2f + (size_t)((3 * 8 + nt) * 32) * 16);
  float bias = biasp[ch];
#pragma unroll
  for (int mt = 0; mt < 8; ++mt) {
    v8f c = {};
    c = wmma_bf16(load_a_frag(ldsA, mt * 16, 68, 0), B0, c);
    c = wmma_bf16(load_a_frag(ldsA, mt * 16, 68, 1), B1, c);
    c = wmma_bf16(load_a_frag(ldsA, mt * 16, 68, 2), B2, c);
    c = wmma_bf16(load_a_frag(ldsA, mt * 16, 68, 3), B3, c);
#pragma unroll
    for (int r = 0; r < 8; ++r) {
      int row = mt * 16 + r + 8 * half;
      atomicAdd(&agg[(long)ldsCol[row] * 128 + ch], c[r] + bias);
    }
  }
}

// Node pass 1: h3 = leakyrelu([x | agg] @ W3 + b3); store bf16 + stats. K = 192.
__global__ __launch_bounds__(256) void node_mlp1(
    const float* __restrict__ x, const float* __restrict__ agg,
    const float* __restrict__ b3, const unsigned short* __restrict__ w3f,
    unsigned short* __restrict__ h3, float* __restrict__ stats) {
  __shared__ __align__(16) unsigned ldsA[128 * 100];
  int tid = threadIdx.x;
  long n0 = (long)blockIdx.x * 128;
  {
    int nr = tid >> 1, hf = tid & 1;
    long node = n0 + nr;
    unsigned* dst = ldsA + nr * 100 + (hf ? 32 : 0);
    if (hf == 0) {
      if (node < NNODES) {
        const float4* src = (const float4*)(x + node * 64);
#pragma unroll
        for (int i = 0; i < 16; ++i) { float4 v = src[i]; dst[2*i] = pack2(v.x, v.y); dst[2*i+1] = pack2(v.z, v.w); }
      } else { for (int i = 0; i < 32; ++i) dst[i] = 0u; }
    } else {
      if (node < NNODES) {
        const float4* src = (const float4*)(agg + node * 128);
#pragma unroll
        for (int i = 0; i < 32; ++i) { float4 v = src[i]; dst[2*i] = pack2(v.x, v.y); dst[2*i+1] = pack2(v.z, v.w); }
      } else { for (int i = 0; i < 64; ++i) dst[i] = 0u; }
    }
  }
  __syncthreads();
  int lane = tid & 31, nt = tid >> 5;
  int n = lane & 15, half = lane >> 4;
  int ch = nt * 16 + n;
  v16bf Bf[6];
#pragma unroll
  for (int kt = 0; kt < 6; ++kt) Bf[kt] = load_b_frag(w3f + (size_t)((kt * 8 + nt) * 32) * 16);
  float bias = b3[ch];
  float ssum = 0.f, ssq = 0.f;
#pragma unroll
  for (int mt = 0; mt < 8; ++mt) {
    v8f c = {};
#pragma unroll
    for (int kt = 0; kt < 6; ++kt)
      c = wmma_bf16(load_a_frag(ldsA, mt * 16, 100, kt), Bf[kt], c);
#pragma unroll
    for (int r = 0; r < 8; ++r) {
      long row = n0 + mt * 16 + r + 8 * half;
      if (row < NNODES) {
        float v = c[r] + bias;
        v = v > 0.f ? v : SLOPE * v;
        ssum += v; ssq += v * v;
        h3[row * 128 + ch] = f2bf(v);
      }
    }
  }
  atomicAdd(&stats[ch], ssum);
  atomicAdd(&stats[128 + ch], ssq);
}

// Node pass 2: out = h3 @ W4' + b4'. Output 64 cols: waves 0-3 nt, waves 4-7 mirror on M.
__global__ __launch_bounds__(256) void node_mlp2(
    const unsigned short* __restrict__ h3, const unsigned short* __restrict__ w4f,
    const float* __restrict__ biasp, float* __restrict__ out) {
  __shared__ __align__(16) unsigned ldsA[128 * 68];
  int tid = threadIdx.x;
  long n0 = (long)blockIdx.x * 128;
  {
    int nr = tid >> 1, hf = tid & 1;
    long node = n0 + nr;
    unsigned* dst = ldsA + nr * 68 + hf * 32;
    if (node < NNODES) {
      const unsigned short* gsrc = h3 + node * 128 + hf * 64;
#pragma unroll
      for (int i = 0; i < 8; ++i) async_copy_b128(gsrc + i * 8, dst + i * 4);
    } else {
      uint4 z = {0u, 0u, 0u, 0u};
      for (int i = 0; i < 8; ++i) ((uint4*)dst)[i] = z;
    }
  }
  wait_async0();
  __syncthreads();
  int lane = tid & 31, w = tid >> 5;
  int nt = w & 3, mbase = (w >> 2) * 4;
  int n = lane & 15, half = lane >> 4;
  int ch = nt * 16 + n;
  v16bf Bf[4];
#pragma unroll
  for (int kt = 0; kt < 4; ++kt) Bf[kt] = load_b_frag(w4f + (size_t)((kt * 4 + nt) * 32) * 16);
  float bias = biasp[ch];
#pragma unroll
  for (int mti = 0; mti < 4; ++mti) {
    int mt = mbase + mti;
    v8f c = {};
#pragma unroll
    for (int kt = 0; kt < 4; ++kt)
      c = wmma_bf16(load_a_frag(ldsA, mt * 16, 68, kt), Bf[kt], c);
#pragma unroll
    for (int r = 0; r < 8; ++r) {
      long row = n0 + mt * 16 + r + 8 * half;
      if (row < NNODES) out[row * 64 + ch] = c[r] + bias;
    }
  }
}

extern "C" void kernel_launch(void* const* d_in, const int* in_sizes, int n_in,
                              void* d_out, int out_size, void* d_ws, size_t ws_size,
                              hipStream_t stream) {
  const float* x   = (const float*)d_in[0];
  const int*   ei  = (const int*)d_in[1];
  const float* ea  = (const float*)d_in[2];
  const float* W1  = (const float*)d_in[5];
  const float* b1  = (const float*)d_in[6];
  const float* g1  = (const float*)d_in[7];
  const float* be1 = (const float*)d_in[8];
  const float* W2  = (const float*)d_in[9];
  const float* b2  = (const float*)d_in[10];
  const float* W3  = (const float*)d_in[11];
  const float* b3  = (const float*)d_in[12];
  const float* g2  = (const float*)d_in[13];
  const float* be2 = (const float*)d_in[14];
  const float* W4  = (const float*)d_in[15];
  const float* b4  = (const float*)d_in[16];
  float* out = (float*)d_out;

  char* ws = (char*)d_ws;
  size_t off = 0;
  auto alloc = [&](size_t bytes) { size_t o = off; off = (off + bytes + 255) & ~(size_t)255; return o; };
  unsigned short* w1f = (unsigned short*)(ws + alloc(128 * 128 * 2));
  unsigned short* w2f = (unsigned short*)(ws + alloc(128 * 128 * 2));
  unsigned short* w3f = (unsigned short*)(ws + alloc(192 * 128 * 2));
  unsigned short* w4f = (unsigned short*)(ws + alloc(128 * 64 * 2));
  float* bias2p = (float*)(ws + alloc(128 * 4));
  float* bias4p = (float*)(ws + alloc(64 * 4));
  float* stats1 = (float*)(ws + alloc(256 * 4));
  float* stats2 = (float*)(ws + alloc(256 * 4));
  float* agg          = (float*)(ws + alloc((size_t)NNODES * 128 * 4));
  unsigned short* h3w = (unsigned short*)(ws + alloc((size_t)NNODES * 128 * 2));
  unsigned short* h1w = (unsigned short*)(ws + alloc((size_t)NEDGES * 128 * 2));

  // Zero accumulators (deterministic per call; harness does not re-poison).
  long aggN = (long)NNODES * 128;
  zero_f32<<<(int)((aggN + 255) / 256), 256, 0, stream>>>(agg, aggN);
  zero_f32<<<1, 256, 0, stream>>>(stats1, 256);
  zero_f32<<<1, 256, 0, stream>>>(stats2, 256);

  prep_weights<<<160, 256, 0, stream>>>(W1, W3, w1f, w3f);

  edge_mlp1<<<NEDGES / 128, 256, 0, stream>>>(x, ei, ea, b1, w1f, h1w, stats1);
  fold_bn<<<1, 128, 0, stream>>>(stats1, (float)NEDGES, g1, be1, W2, b2, 128, 8, w2f, bias2p);
  edge_mlp2<<<NEDGES / 128, 256, 0, stream>>>(h1w, ei, w2f, bias2p, agg);

  int nodeBlocks = (NNODES + 127) / 128;
  node_mlp1<<<nodeBlocks, 256, 0, stream>>>(x, agg, b3, w3f, h3w, stats2);
  fold_bn<<<1, 128, 0, stream>>>(stats2, (float)NNODES, g2, be2, W4, b4, 64, 4, w4f, bias4p);
  node_mlp2<<<nodeBlocks, 256, 0, stream>>>(h3w, w4f, bias4p, out);
}
